// GraphAttentionLayer_32461362823922
// MI455X (gfx1250) — matrix operand order repaired and checked by reading, not compiled, not run on hardware
//
#include <hip/hip_runtime.h>

typedef __attribute__((ext_vector_type(16))) _Float16     v16h;
typedef __attribute__((ext_vector_type(8)))  float        v8f;
typedef __attribute__((ext_vector_type(2)))  float        v2f;
typedef __attribute__((ext_vector_type(4)))  unsigned int v4u;
typedef __attribute__((ext_vector_type(8)))  int          v8i;
typedef __attribute__((ext_vector_type(4)))  int          v4i;

// 32-bit LDS-address-space pointer for real ds_load codegen
typedef const __attribute__((address_space(3))) v16h* lds_v16h_cptr;

#define N_NODES 8192
#define FDIM    128
#define NEG_INF_F (-1.0e9f)

// TDM-staged B slice: 128 rows x 32 f16 cols, padded 64B->80B pitch so
// ds_load_b128 rows are 16B-aligned and bank-conflict free (m*20 mod 64).
#define BUF_PITCH 80
#define BUF_BYTES (128 * BUF_PITCH)   // 10240 B per buffer

// ---------------------------------------------------------------------------
// Prep: Wh = h @ W via v_wmma_f32_16x16x4_f32 (full f32 precision),
// emit WhT as fp16 (transposed, contiguous B tiles), s_key = Wh@a1, s_qry = Wh@a2.
// One wave per 16-row tile of Wh.
// ---------------------------------------------------------------------------
__global__ __launch_bounds__(32) void gat_prep(
    const float* __restrict__ h, const float* __restrict__ W,
    const float* __restrict__ a,
    _Float16* __restrict__ WhT, float* __restrict__ skey, float* __restrict__ sqry)
{
  const int lane = threadIdx.x & 31;
  const int m    = lane & 15;
  const int kh   = lane >> 4;
  const int row0 = blockIdx.x * 16;

  v8f c[8] = {};

  for (int k = 0; k < FDIM; k += 4) {
    const float* hp = h + (size_t)(row0 + m) * FDIM + (k + kh * 2);
    v2f av = *(const v2f*)hp;
    #pragma unroll
    for (int t = 0; t < 8; ++t) {
      const int n = t * 16 + m;
      v2f bv;
      bv.x = W[(size_t)(k + kh * 2 + 0) * FDIM + n];
      bv.y = W[(size_t)(k + kh * 2 + 1) * FDIM + n];
      c[t] = __builtin_amdgcn_wmma_f32_16x16x4_f32(
          false, av, false, bv, (short)0, c[t], false, false);
    }
  }

  float skp[8], sqp[8];
  #pragma unroll
  for (int r = 0; r < 8; ++r) { skp[r] = 0.0f; sqp[r] = 0.0f; }

  #pragma unroll
  for (int t = 0; t < 8; ++t) {
    const int n = t * 16 + m;
    const float a1 = a[n];
    const float a2 = a[FDIM + n];
    #pragma unroll
    for (int r = 0; r < 8; ++r) {
      const float v = c[t][r];
      WhT[(size_t)n * N_NODES + (row0 + kh * 8 + r)] = (_Float16)v;
      skp[r] += v * a1;
      sqp[r] += v * a2;
    }
  }
  #pragma unroll
  for (int d = 1; d < 16; d <<= 1) {
    #pragma unroll
    for (int r = 0; r < 8; ++r) {
      skp[r] += __shfl_xor(skp[r], d, 32);
      sqp[r] += __shfl_xor(sqp[r], d, 32);
    }
  }
  if (m == 0) {
    #pragma unroll
    for (int r = 0; r < 8; ++r) {
      skey[row0 + kh * 8 + r] = skp[r];
      sqry[row0 + kh * 8 + r] = sqp[r];
    }
  }
}

// ---------------------------------------------------------------------------
// TDM: DMA WhT[:, j..j+32) (128 rows x 64B) into LDS at lds_off, with
// 16B padding inserted every 64B (row pitch 80B).  D# per ISA 08 §8.
// ---------------------------------------------------------------------------
__device__ __forceinline__ void tdm_stage_b_slice(const _Float16* gsrc, unsigned lds_off)
{
  const unsigned long long ga = (unsigned long long)(size_t)gsrc;

  v4u g0;
  g0.x = 1u;                                   // count=1, user descriptor
  g0.y = lds_off;                              // lds_addr (bytes)
  g0.z = (unsigned)(ga & 0xffffffffu);         // global_addr[31:0]
  g0.w = (unsigned)((ga >> 32) & 0x01ffffffu)  // global_addr[56:32]
         | 0x80000000u;                        // type = 2 ("image")

  v8i g1;
  g1[0] = (1 << 16)      // data_size = 2 bytes
        | (1 << 20)      // pad_enable
        | (3 << 22)      // pad_interval: every 16 DWORDs (64B)
        | (3 << 25);     // pad_amount: 4 DWORDs (16B) -> pitch 80B
  g1[1] = (N_NODES & 0xffff) << 16;            // tensor_dim0 lo16 (8192)
  g1[2] = (N_NODES >> 16) | (FDIM << 16);      // tensor_dim0 hi16, tensor_dim1 lo16 (128)
  g1[3] = (32 << 16);                          // tensor_dim1 hi16=0, tile_dim0 = 32
  g1[4] = FDIM;                                // tile_dim1 = 128, tile_dim2 = 0
  g1[5] = N_NODES;                             // tensor_dim0_stride lo32 = 8192
  g1[6] = 0;                                   // stride hi16, dim1_stride lo16
  g1[7] = 0;

  v4i gz = {0, 0, 0, 0};                       // groups 2/3: unused (2D tensor)
#if defined(__clang_major__) && __clang_major__ >= 23
  v8i gz8 = {0, 0, 0, 0, 0, 0, 0, 0};
  __builtin_amdgcn_tensor_load_to_lds(g0, g1, gz, gz, gz8, 0);
#else
  __builtin_amdgcn_tensor_load_to_lds(g0, g1, gz, gz, 0);
#endif
}

// ---------------------------------------------------------------------------
// Fused masked-softmax attention + h_prime = attention @ Wh + ELU.
// One wave per 16-row tile; online softmax over 32-col chunks; B operand
// double-buffered in LDS via TDM; 8x v_wmma_f32_16x16x32_f16 per chunk.
// ---------------------------------------------------------------------------
__device__ __forceinline__ float gat_score(float sq, float sk, int ad) {
  float v = sq + sk;
  v = (v > 0.0f) ? v : 0.2f * v;   // leaky_relu(0.2)
  return (ad > 0) ? v : NEG_INF_F;
}

__global__ __launch_bounds__(32) void gat_attn(
    const int* __restrict__ adj, const _Float16* __restrict__ WhT,
    const float* __restrict__ skey, const float* __restrict__ sqry,
    float* __restrict__ out)
{
  __shared__ __attribute__((aligned(128))) char ldsbuf[2][BUF_BYTES];

  const int lane = threadIdx.x & 31;
  const int m    = lane & 15;
  const int kh   = lane >> 4;
  const int row0 = blockIdx.x * 16;
  const int row  = row0 + m;

  const float sq = sqry[row];
  const int* __restrict__ ar = adj + (size_t)row * N_NODES;

  // low 32 bits of the flat LDS address == workgroup-relative LDS byte offset
  const unsigned lds_base = (unsigned)(size_t)(void*)&ldsbuf[0][0];

  float mrun = -__builtin_inff();
  float lrun = 0.0f;
  v8f acc[8] = {};

  // prologue: stage chunk 0 into buffer 0
  tdm_stage_b_slice(WhT, lds_base);

  for (int j = 0; j < N_NODES; j += 32) {
    // stage next chunk into the other buffer, then wait for current chunk
    if (j + 32 < N_NODES) {
      tdm_stage_b_slice(WhT + (j + 32), lds_base + (((j >> 5) + 1) & 1) * BUF_BYTES);
      __builtin_amdgcn_s_wait_tensorcnt(1);
    } else {
      __builtin_amdgcn_s_wait_tensorcnt(0);
    }

    // Laundered 32-bit LDS offset for this lane's B rows: keeps the ds reads
    // below the tensorcnt wait and opaque to alias analysis, while staying in
    // address space 3 so they lower to ds_load_b128 (not flat).
    unsigned ba = lds_base + ((j >> 5) & 1) * BUF_BYTES
                + (unsigned)(m * BUF_PITCH + kh * 32);
    asm volatile("" : "+v"(ba) : : "memory");

    // ---- scores for this lane's 16 A-matrix K positions ----
    const int c0 = j + kh * 16;        // A elements 0..7
    const int c1 = j + kh * 16 + 8;    // A elements 8..15

    __builtin_prefetch(ar + j + 256, 0, 0);

    const int4   A0 = *(const int4*)(ar + c0);
    const int4   A1 = *(const int4*)(ar + c0 + 4);
    const int4   A2 = *(const int4*)(ar + c1);
    const int4   A3 = *(const int4*)(ar + c1 + 4);
    const float4 K0 = *(const float4*)(skey + c0);
    const float4 K1 = *(const float4*)(skey + c0 + 4);
    const float4 K2 = *(const float4*)(skey + c1);
    const float4 K3 = *(const float4*)(skey + c1 + 4);

    float s[16];
    s[ 0] = gat_score(sq, K0.x, A0.x); s[ 1] = gat_score(sq, K0.y, A0.y);
    s[ 2] = gat_score(sq, K0.z, A0.z); s[ 3] = gat_score(sq, K0.w, A0.w);
    s[ 4] = gat_score(sq, K1.x, A1.x); s[ 5] = gat_score(sq, K1.y, A1.y);
    s[ 6] = gat_score(sq, K1.z, A1.z); s[ 7] = gat_score(sq, K1.w, A1.w);
    s[ 8] = gat_score(sq, K2.x, A2.x); s[ 9] = gat_score(sq, K2.y, A2.y);
    s[10] = gat_score(sq, K2.z, A2.z); s[11] = gat_score(sq, K2.w, A2.w);
    s[12] = gat_score(sq, K3.x, A3.x); s[13] = gat_score(sq, K3.y, A3.y);
    s[14] = gat_score(sq, K3.z, A3.z); s[15] = gat_score(sq, K3.w, A3.w);

    float cm = s[0];
    #pragma unroll
    for (int i = 1; i < 16; ++i) cm = fmaxf(cm, s[i]);
    cm = fmaxf(cm, __shfl_xor(cm, 16, 32));   // row max over the 32-col chunk

    // rare wave-uniform rescale path (EXEC stays all-ones for the WMMAs)
    if (__any(cm > mrun)) {
      const float mnew  = fmaxf(mrun, cm);
      const float alpha = __expf(mrun - mnew);
      mrun = mnew;
      lrun *= alpha;
      #pragma unroll
      for (int r = 0; r < 8; ++r) {
        const float arr = __shfl(alpha, r + kh * 8, 32);
        #pragma unroll
        for (int t = 0; t < 8; ++t) acc[t][r] *= arr;
      }
    }

    float p[16];
    float psum = 0.0f;
    #pragma unroll
    for (int i = 0; i < 16; ++i) { p[i] = __expf(s[i] - mrun); psum += p[i]; }
    lrun += psum + __shfl_xor(psum, 16, 32);

    v16h Av;
    #pragma unroll
    for (int i = 0; i < 16; ++i) Av[i] = (_Float16)p[i];

    // B tiles from LDS via ds_load_b128: staged row (t*16+m), cols kh*16..+16
    #pragma unroll
    for (int t = 0; t < 8; ++t) {
      const v16h Bv = *(lds_v16h_cptr)(ba + (unsigned)(t * 16 * BUF_PITCH));
      acc[t] = __builtin_amdgcn_wmma_f32_16x16x32_f16(
          false, Av, false, Bv, (short)0, acc[t], false, false);
    }
  }

  // normalize, ELU, store
  #pragma unroll
  for (int r = 0; r < 8; ++r) {
    const float lr  = __shfl(lrun, r + kh * 8, 32);
    const float inv = 1.0f / lr;
    const int orow  = row0 + kh * 8 + r;
    #pragma unroll
    for (int t = 0; t < 8; ++t) {
      float v = acc[t][r] * inv;
      v = (v > 0.0f) ? v : (__expf(v) - 1.0f);  // elu
      out[(size_t)orow * FDIM + t * 16 + m] = v;
    }
  }
}

extern "C" void kernel_launch(void* const* d_in, const int* in_sizes, int n_in,
                              void* d_out, int out_size, void* d_ws, size_t ws_size,
                              hipStream_t stream) {
  const float* h   = (const float*)d_in[0];   // (8192,128) f32
  const int*   adj = (const int*)d_in[1];     // (8192,8192) i32
  const float* W   = (const float*)d_in[2];   // (128,128) f32
  const float* a   = (const float*)d_in[3];   // (256,) f32
  float* out = (float*)d_out;                 // (8192,128) f32

  char* ws = (char*)d_ws;
  _Float16* WhT  = (_Float16*)ws;                               // 2 MB
  float*    skey = (float*)(ws + (size_t)FDIM * N_NODES * 2);   // 32 KB
  float*    sqry = skey + N_NODES;                              // 32 KB

  gat_prep<<<N_NODES / 16, 32, 0, stream>>>(h, W, a, WhT, skey, sqry);
  gat_attn<<<N_NODES / 16, 32, 0, stream>>>(adj, WhT, skey, sqry, out);
}
